// PlainGNN_77360950936037
// MI455X (gfx1250) — matrix operand order repaired and checked by reading, not compiled, not run on hardware
//
#include <hip/hip_runtime.h>
#include <hip/hip_bf16.h>
#include <math.h>

// ---------------------------------------------------------------------------
// PlainGNN (GAT-like) for MI455X / gfx1250, wave32.
//   Q,K,V projections: v_wmma_f32_16x16x32_bf16 (f32 accumulate)
//   Edge phase: one wave per edge, float4/lane coalesced gathers, f32 atomics
// ---------------------------------------------------------------------------

#define IN_F       128
#define OUT_TOTAL  128
#define HEADS      4
#define D_K        32

typedef __bf16 bf16_t;
typedef bf16_t v16bf __attribute__((ext_vector_type(16)));
typedef float  v8f   __attribute__((ext_vector_type(8)));

// -------------------- init kernels --------------------
__global__ void init_zero_f32(float* __restrict__ p, int n) {
    int t = blockIdx.x * blockDim.x + threadIdx.x;
    if (t < n) p[t] = 0.0f;
}

__global__ void init_stats(float* __restrict__ smax, float* __restrict__ ssum, int n) {
    int t = blockIdx.x * blockDim.x + threadIdx.x;
    if (t < n) { smax[t] = -INFINITY; ssum[t] = 0.0f; }
}

// -------------------- QKV projection via WMMA --------------------
// One wave handles a 16-node x 16-outfeature tile for all three matrices.
// Grid: node_tiles * 8 out_tiles waves, 8 waves (256 threads) per block.
__global__ __launch_bounds__(256) void qkv_proj_wmma(
    const float* __restrict__ x,
    const float* __restrict__ Wq, const float* __restrict__ Wk, const float* __restrict__ Wv,
    float* __restrict__ Q, float* __restrict__ K, float* __restrict__ V, int N)
{
    const int wave = (blockIdx.x * blockDim.x + threadIdx.x) >> 5;
    const int lane = threadIdx.x & 31;
    const int nt   = wave >> 3;   // node tile index
    const int ot   = wave & 7;    // out-feature tile index (8 * 16 = 128)
    const int half = lane >> 4;   // lane group 0/1
    const int lrow = lane & 15;

    // A operand: row = node; clamp tail reads to stay in-bounds (store guarded).
    int arow = nt * 16 + lrow;
    int arow_c = arow < N ? arow : (N - 1);
    const float* xr = x + (size_t)arow_c * IN_F;
    // B operand: row of W = output feature.
    const int wbase = (ot * 16 + lrow) * IN_F;

    v8f accq = {}; v8f acck = {}; v8f accv = {};

    #pragma unroll
    for (int kb = 0; kb < IN_F; kb += 32) {
        // ---- A (x tile), contiguous k order (matches B):
        //  lanes 0-15 : K = kb+0..15
        //  lanes 16-31: K = kb+16..31
        const int ka0 = kb + half * 16;
        const int ka1 = kb + half * 16 + 8;
        v16bf a;
        #pragma unroll
        for (int i = 0; i < 8; ++i) {
            a[i]     = (bf16_t)xr[ka0 + i];
            a[8 + i] = (bf16_t)xr[ka1 + i];
        }
        // ---- B (W^T tile): column n = W row n; contiguous k order (same as A):
        //  lanes 0-15 hold K = kb+0..15, lanes 16-31 hold K = kb+16..31
        const int kb0 = wbase + kb + half * 16;
        v16bf bq, bk, bv;
        #pragma unroll
        for (int i = 0; i < 16; ++i) {
            bq[i] = (bf16_t)Wq[kb0 + i];
            bk[i] = (bf16_t)Wk[kb0 + i];
            bv[i] = (bf16_t)Wv[kb0 + i];
        }
        accq = __builtin_amdgcn_wmma_f32_16x16x32_bf16(false, a, false, bq, (short)0, accq, false, false);
        acck = __builtin_amdgcn_wmma_f32_16x16x32_bf16(false, a, false, bk, (short)0, acck, false, false);
        accv = __builtin_amdgcn_wmma_f32_16x16x32_bf16(false, a, false, bv, (short)0, accv, false, false);
    }

    // D layout: acc[r] -> M = r + 8*half, N = lrow
    const int mrow = nt * 16 + half * 8;
    #pragma unroll
    for (int r = 0; r < 8; ++r) {
        int row = mrow + r;
        if (row < N) {
            size_t o = (size_t)row * OUT_TOTAL + ot * 16 + lrow;
            Q[o] = accq[r];
            K[o] = acck[r];
            V[o] = accv[r];
        }
    }
}

// -------------------- float atomic max via int bit tricks --------------------
__device__ __forceinline__ void atomicMaxF(float* addr, float val) {
    if (!(val < 0.0f)) atomicMax((int*)addr, __float_as_int(val));
    else               atomicMin((unsigned int*)addr, __float_as_uint(val));
}

// -------------------- pass 1: edge scores + segment max --------------------
// One wave per edge; lane l handles float4 at offset 4*l -> head h = l>>3.
__global__ __launch_bounds__(256) void edge_scores(
    const float* __restrict__ Km, const float* __restrict__ Qm,
    const int* __restrict__ row, const int* __restrict__ col,
    const float* __restrict__ Dw,
    float* __restrict__ scores, float* __restrict__ smax, int E)
{
    const int e = (blockIdx.x * blockDim.x + threadIdx.x) >> 5;
    if (e >= E) return;
    const int lane = threadIdx.x & 31;
    const int r = row[e];
    const int c = col[e];

    const float4 kv = *(const float4*)(Km + (size_t)r * OUT_TOTAL + lane * 4);
    const float4 qv = *(const float4*)(Qm + (size_t)c * OUT_TOTAL + lane * 4);
    const float4 dv = *(const float4*)(Dw + lane * 4);

    float s;
    if (r == c) {
        // self edge: K . D . Q(row)  (col == row so qv is Q[row])
        s = kv.x * dv.x * qv.x + kv.y * dv.y * qv.y +
            kv.z * dv.z * qv.z + kv.w * dv.w * qv.w;
    } else {
        float dx = kv.x - qv.x, dy = kv.y - qv.y, dz = kv.z - qv.z, dw = kv.w - qv.w;
        s = dx * dv.x * dx + dy * dv.y * dy + dz * dv.z * dz + dw * dv.w * dw;
    }
    // reduce within each 8-lane head group
    s += __shfl_xor(s, 1);
    s += __shfl_xor(s, 2);
    s += __shfl_xor(s, 4);

    if ((lane & 7) == 0) {
        const int h = lane >> 3;
        const float scale = 0.17677669529663687f;   // 1/sqrt(32)
        s *= scale;
        s = (s > 0.0f) ? s : 0.2f * s;              // leaky_relu 0.2
        scores[(size_t)e * HEADS + h] = s;
        atomicMaxF(&smax[r * HEADS + h], s);
    }
}

// -------------------- pass 2: exp + segment sum --------------------
__global__ __launch_bounds__(256) void edge_exp(
    const int* __restrict__ row, float* __restrict__ scores,
    const float* __restrict__ smax, float* __restrict__ ssum, int EH)
{
    int t = blockIdx.x * blockDim.x + threadIdx.x;
    if (t >= EH) return;
    const int e = t >> 2;
    const int h = t & 3;
    const int r = row[e];
    float ex = __expf(scores[t] - smax[r * HEADS + h]);
    scores[t] = ex;                       // overwrite score with exp term
    atomicAdd(&ssum[r * HEADS + h], ex);
}

// -------------------- pass 3: alpha * V scatter-add --------------------
__global__ __launch_bounds__(256) void edge_aggregate(
    const float* __restrict__ Vm,
    const int* __restrict__ row, const int* __restrict__ col,
    const float* __restrict__ ex, const float* __restrict__ ssum,
    float* __restrict__ out, int E)
{
    const int e = (blockIdx.x * blockDim.x + threadIdx.x) >> 5;
    if (e >= E) return;
    const int lane = threadIdx.x & 31;
    const int r = row[e];
    const int c = col[e];
    const int h = lane >> 3;

    const float alpha = ex[(size_t)e * HEADS + h] / (ssum[r * HEADS + h] + 1e-16f);
    const float4 vv = *(const float4*)(Vm + (size_t)c * OUT_TOTAL + lane * 4);
    float* o = out + (size_t)r * OUT_TOTAL + lane * 4;
    atomicAdd(o + 0, alpha * vv.x);
    atomicAdd(o + 1, alpha * vv.y);
    atomicAdd(o + 2, alpha * vv.z);
    atomicAdd(o + 3, alpha * vv.w);
}

// -------------------- launcher --------------------
extern "C" void kernel_launch(void* const* d_in, const int* in_sizes, int n_in,
                              void* d_out, int out_size, void* d_ws, size_t ws_size,
                              hipStream_t stream)
{
    const float* x  = (const float*)d_in[0];
    const float* Wq = (const float*)d_in[1];
    const float* Wk = (const float*)d_in[2];
    const float* Wv = (const float*)d_in[3];
    const float* Dw = (const float*)d_in[4];
    const int*   ei = (const int*)d_in[5];

    const int N = in_sizes[0] / IN_F;       // 100000
    const int E = in_sizes[5] / 2;          // 1600000
    const int* row = ei;
    const int* col = ei + E;

    // workspace layout (floats)
    float* ws = (float*)d_ws;
    size_t nQ = (size_t)N * OUT_TOTAL;
    float* Qm     = ws;
    float* Km     = Qm + nQ;
    float* Vm     = Km + nQ;
    float* scores = Vm + nQ;                     // E*HEADS, reused as exp terms
    float* smax   = scores + (size_t)E * HEADS;  // N*HEADS
    float* ssum   = smax + (size_t)N * HEADS;    // N*HEADS
    (void)ws_size; (void)n_in; (void)out_size;

    float* out = (float*)d_out;

    // init: out = 0, smax = -inf, ssum = 0
    {
        int n = N * OUT_TOTAL;
        init_zero_f32<<<(n + 255) / 256, 256, 0, stream>>>(out, n);
        int m = N * HEADS;
        init_stats<<<(m + 255) / 256, 256, 0, stream>>>(smax, ssum, m);
    }

    // QKV projections via WMMA: (node_tiles * 8) waves, 8 waves per block
    {
        int node_tiles = (N + 15) / 16;
        int waves = node_tiles * 8;
        int blocks = (waves + 7) / 8;            // 256 threads = 8 waves
        qkv_proj_wmma<<<blocks, 256, 0, stream>>>(x, Wq, Wk, Wv, Qm, Km, Vm, N);
    }

    // edge scores + segment max (one wave per edge, 8 edges per block)
    edge_scores<<<(E + 7) / 8, 256, 0, stream>>>(Km, Qm, row, col, Dw, scores, smax, E);

    // exp + segment sum (one thread per edge*head)
    {
        int EH = E * HEADS;
        edge_exp<<<(EH + 255) / 256, 256, 0, stream>>>(row, scores, smax, ssum, EH);
    }

    // aggregate alpha * V[col] into out[row]
    edge_aggregate<<<(E + 7) / 8, 256, 0, stream>>>(Vm, row, col, scores, ssum, out, E);
}
